// LightGCN_27917287424334
// MI455X (gfx1250) — matrix operand order repaired and checked by reading, not compiled, not run on hardware
//
#include <hip/hip_runtime.h>
#include <hip/hip_bf16.h>
#include <stddef.h>

// ---------------- problem constants (match reference) ----------------
#define NUM_USERS 100000
#define NUM_ITEMS 50000
#define EMBED_DIM 64
#define N_LAYERS  3
#define BATCH     1024
#define N_NODES   (NUM_USERS + NUM_ITEMS)      // 150000
#define NODE_FLOATS ((size_t)N_NODES * EMBED_DIM)   // 9.6M floats
#define USER_FLOATS ((size_t)NUM_USERS * EMBED_DIM) // 6.4M floats

typedef __attribute__((ext_vector_type(16))) _Float16 v16h;
typedef __attribute__((ext_vector_type(8)))  _Float16 v8h;
typedef __attribute__((ext_vector_type(8)))  float    v8f;

// ---------------------------------------------------------------------
// Kernel 1: emb = acc = concat(user_emb, item_emb); emb_next = 0
// float4-vectorized over 9.6M floats (2.4M float4s).
// ---------------------------------------------------------------------
__global__ void lg_init(const float* __restrict__ user_emb,
                        const float* __restrict__ item_emb,
                        float* __restrict__ emb,
                        float* __restrict__ acc,
                        float* __restrict__ emb_next) {
    size_t i = (size_t)blockIdx.x * blockDim.x + threadIdx.x;   // float4 index
    size_t nf4 = NODE_FLOATS / 4;
    if (i >= nf4) return;
    float4 v;
    if (i * 4 < USER_FLOATS) v = ((const float4*)user_emb)[i];
    else                     v = ((const float4*)item_emb)[i - USER_FLOATS / 4];
    ((float4*)emb)[i] = v;
    ((float4*)acc)[i] = v;
    float4 z; z.x = 0.f; z.y = 0.f; z.z = 0.f; z.w = 0.f;
    ((float4*)emb_next)[i] = z;
}

// ---------------------------------------------------------------------
// Kernel 2: edge scatter. One wave32 per edge; each lane handles 2 dims.
// emb[dst]*val scatter-added into emb_next[src] via global_atomic_add_f32.
// Node table (38.4MB) is L2-resident on MI455X (192MB L2); only the edge
// lists (48MB/layer) stream from HBM, and those also stay L2-resident
// across the 3 layers.
// ---------------------------------------------------------------------
__global__ void lg_scatter(const float* __restrict__ emb,
                           float* __restrict__ emb_next,
                           const float* __restrict__ edge_val,
                           const int*   __restrict__ edge_src,
                           const int*   __restrict__ edge_dst,
                           int n_edges) {
    int lane = threadIdx.x & 31;
    int e = (int)(((size_t)blockIdx.x * blockDim.x + threadIdx.x) >> 5);
    if (e >= n_edges) return;
    int   src = edge_src[e];
    int   dst = edge_dst[e];
    float v   = edge_val[e];
    float2 m = ((const float2*)(emb + (size_t)dst * EMBED_DIM))[lane];
    float* outp = emb_next + (size_t)src * EMBED_DIM + lane * 2;
    atomicAdd(outp,     m.x * v);
    atomicAdd(outp + 1, m.y * v);
}

// ---------------------------------------------------------------------
// Kernel 3: acc += emb_next; emb = emb_next; emb_next = 0 (ready for next
// layer). float4-vectorized.
// ---------------------------------------------------------------------
__global__ void lg_update(float* __restrict__ acc,
                          float* __restrict__ emb,
                          float* __restrict__ emb_next) {
    size_t i = (size_t)blockIdx.x * blockDim.x + threadIdx.x;
    size_t nf4 = NODE_FLOATS / 4;
    if (i >= nf4) return;
    float4 v = ((float4*)emb_next)[i];
    float4 a = ((float4*)acc)[i];
    a.x += v.x; a.y += v.y; a.z += v.z; a.w += v.w;
    ((float4*)acc)[i] = a;
    ((float4*)emb)[i] = v;
    float4 z; z.x = 0.f; z.y = 0.f; z.z = 0.f; z.w = 0.f;
    ((float4*)emb_next)[i] = z;
}

// ---------------------------------------------------------------------
// Kernel 4a: gather the batch users and pack A to f16, folding in the
// (1/4)*(1/4) light_out scaling -> * 1/16.
// ---------------------------------------------------------------------
__global__ void lg_pack_a(const float* __restrict__ acc,
                          const int*   __restrict__ users,
                          _Float16*    __restrict__ A16) {
    int tid = blockIdx.x * blockDim.x + threadIdx.x;
    if (tid >= BATCH * EMBED_DIM) return;
    int b = tid >> 6, k = tid & 63;
    int u = users[b];
    A16[tid] = (_Float16)(acc[(size_t)u * EMBED_DIM + k] * 0.0625f);
}

// Kernel 4b: pack item rows (acc rows NUM_USERS..) to f16, unscaled.
__global__ void lg_pack_b(const float* __restrict__ acc,
                          _Float16*    __restrict__ B16) {
    size_t tid = (size_t)blockIdx.x * blockDim.x + threadIdx.x;
    if (tid >= (size_t)NUM_ITEMS * EMBED_DIM) return;
    B16[tid] = (_Float16)acc[USER_FLOATS + tid];
}

// ---------------------------------------------------------------------
// Kernel 5: ratings = A(1024x64,f16) @ B(50000x64,f16)^T via
// v_wmma_f32_16x16x32_f16. Block = 8 waves; one 16-row M tile per block,
// 8*64 = 512 columns; each wave does 4 N-tiles (16x64) reusing its two A
// fragments (K=0..31, 32..63).
//
// Pipelining: all 8 B-fragment b128 loads are issued first (row index
// branchlessly clamped for the ragged last block, so no EXEC divergence
// ever reaches a WMMA), then the 8 WMMAs run back-to-back, then results
// are written with NON-TEMPORAL stores: the 204.8MB output exceeds the
// 192MB L2, so NT stores keep the streamed output from evicting the
// 6.4MB B16 matrix that all 64 M-tile passes re-read.
// ---------------------------------------------------------------------
__global__ __launch_bounds__(256) void lg_gemm_wmma(
        const _Float16* __restrict__ A,   // 1024 x 64 row-major
        const _Float16* __restrict__ B,   // 50000 x 64 row-major (items)
        float* __restrict__ out) {        // 1024 x 50000 row-major
    const int lane = threadIdx.x & 31;
    const int wave = threadIdx.x >> 5;
    const int grp  = lane >> 4;           // lane group 0/1
    const int lr   = lane & 15;
    const int m0     = blockIdx.y * 16;
    const int n_base = blockIdx.x * (8 * 64) + wave * 64;

    // --- A fragments: 16x32 f16 tile, contiguous K order (k = 16*grp + s), same as B:
    // lanes 0-15: row M=lr, halves 0..15 = K 0..15
    // lanes 16-31: same rows, K offset +16 (i.e. 16..31)
    v16h afrag[2];
    const _Float16* arow = A + (size_t)(m0 + lr) * EMBED_DIM;
#pragma unroll
    for (int kg = 0; kg < 2; ++kg) {
        const _Float16* p = arow + 32 * kg + 16 * grp;
        v8h lo = *(const v8h*)(p);
        v8h hi = *(const v8h*)(p + 8);
        v16h a;
#pragma unroll
        for (int i = 0; i < 8; ++i) { a[i] = lo[i]; a[i + 8] = hi[i]; }
        afrag[kg] = a;
    }

    // --- Preload all B fragments (one clause of b128 loads).
    // B fragment: 32x16, lane = column; lanes 0-15 hold K 0..15, lanes
    // 16-31 hold K 16..31, halves K-ascending -> 32B contiguous per lane.
    v16h bfrag[4][2];
#pragma unroll
    for (int t = 0; t < 4; ++t) {
        int nr = n_base + t * 16 + lr;
        nr = (nr < NUM_ITEMS) ? nr : (NUM_ITEMS - 1);   // branchless clamp
        const _Float16* brow = B + (size_t)nr * EMBED_DIM + 16 * grp;
        bfrag[t][0] = *(const v16h*)(brow);        // K 0..31 slice
        bfrag[t][1] = *(const v16h*)(brow + 32);   // K 32..63 slice
    }

    // --- 8 WMMAs back-to-back.
    v8f cfrag[4];
#pragma unroll
    for (int t = 0; t < 4; ++t) {
        v8f c = {};
        c = __builtin_amdgcn_wmma_f32_16x16x32_f16(
                false, afrag[0], false, bfrag[t][0], (short)0, c, false, false);
        c = __builtin_amdgcn_wmma_f32_16x16x32_f16(
                false, afrag[1], false, bfrag[t][1], (short)0, c, false, false);
        cfrag[t] = c;
    }

    // --- Non-temporal stores; guard ragged tiles (wave-uniform test).
    // C/D layout: vgpr r, lane -> (M = r + 8*grp, N = lr)
#pragma unroll
    for (int t = 0; t < 4; ++t) {
        int n0 = n_base + t * 16;
        if (n0 < NUM_ITEMS) {
            float* orow = out + (size_t)(m0 + 8 * grp) * NUM_ITEMS + n0 + lr;
#pragma unroll
            for (int r = 0; r < 8; ++r)
                __builtin_nontemporal_store(cfrag[t][r], orow + (size_t)r * NUM_ITEMS);
        }
    }
}

// ---------------------------------------------------------------------
// Host-side orchestration (graph-capture safe: only kernel launches).
// ---------------------------------------------------------------------
extern "C" void kernel_launch(void* const* d_in, const int* in_sizes, int n_in,
                              void* d_out, int out_size, void* d_ws, size_t ws_size,
                              hipStream_t stream) {
    const float* user_emb = (const float*)d_in[0];
    const float* item_emb = (const float*)d_in[1];
    const float* edge_val = (const float*)d_in[2];
    const int*   edge_src = (const int*)  d_in[3];
    const int*   edge_dst = (const int*)  d_in[4];
    const int*   users    = (const int*)  d_in[5];
    const int    n_edges  = in_sizes[2];

    // workspace layout (bytes, 256-aligned)
    const size_t EMB_BYTES = NODE_FLOATS * sizeof(float);     // 38,400,000
    char* ws = (char*)d_ws;
    float*    emb      = (float*)(ws);
    float*    emb_next = (float*)(ws + EMB_BYTES);
    float*    acc      = (float*)(ws + 2 * EMB_BYTES);
    _Float16* A16      = (_Float16*)(ws + 3 * EMB_BYTES);
    _Float16* B16      = (_Float16*)(ws + 3 * EMB_BYTES
                                       + (size_t)BATCH * EMBED_DIM * 2);

    float* out = (float*)d_out;

    const int TB = 256;
    // 1) init
    {
        int n4 = (int)(NODE_FLOATS / 4);
        lg_init<<<(n4 + TB - 1) / TB, TB, 0, stream>>>(user_emb, item_emb,
                                                       emb, acc, emb_next);
    }
    // 2) three propagation layers
    for (int layer = 0; layer < N_LAYERS; ++layer) {
        int blocks = (int)(((size_t)n_edges * 32 + TB - 1) / TB); // wave/edge
        lg_scatter<<<blocks, TB, 0, stream>>>(
            emb, emb_next, edge_val, edge_src, edge_dst, n_edges);
        int n4 = (int)(NODE_FLOATS / 4);
        lg_update<<<(n4 + TB - 1) / TB, TB, 0, stream>>>(acc, emb, emb_next);
    }
    // 3) pack f16 operands
    lg_pack_a<<<(BATCH * EMBED_DIM + TB - 1) / TB, TB, 0, stream>>>(acc, users, A16);
    {
        int n = NUM_ITEMS * EMBED_DIM;
        lg_pack_b<<<(n + TB - 1) / TB, TB, 0, stream>>>(acc, B16);
    }
    // 4) WMMA GEMM: grid.x covers 512 cols/block, grid.y = 1024/16 M tiles
    {
        dim3 grid((NUM_ITEMS + 512 - 1) / 512, BATCH / 16);
        lg_gemm_wmma<<<grid, TB, 0, stream>>>(A16, B16, out);
    }
}
